// SPELL_54271206752661
// MI455X (gfx1250) — matrix operand order repaired and checked by reading, not compiled, not run on hardware
//
#include <hip/hip_runtime.h>
#include <hip/hip_bf16.h>

// ---------------------------------------------------------------------------
// Problem constants (from the reference)
// ---------------------------------------------------------------------------
constexpr int N    = 4096;
constexpr int E    = 262144;
constexpr int FD   = 1024;
constexpr int DL   = 272;
constexpr int DC   = 512;
constexpr int PROJ = 64;
constexpr int EMB  = 640;
constexpr int C1   = 64;
constexpr int C2   = 64;

typedef __attribute__((ext_vector_type(16))) _Float16 v16h;
typedef __attribute__((ext_vector_type(8)))  float    v8f;

union hpair { _Float16 h[2]; unsigned u; };

// ---------------------------------------------------------------------------
// Order-preserving encode for float atomic-max via uint atomicMax
// ---------------------------------------------------------------------------
__device__ __forceinline__ unsigned encF(float f) {
    unsigned u = __float_as_uint(f);
    return (u & 0x80000000u) ? ~u : (u | 0x80000000u);
}
__device__ __forceinline__ float decF(unsigned u) {
    unsigned b = (u & 0x80000000u) ? (u & 0x7FFFFFFFu) : ~u;
    return __uint_as_float(b);
}
// encF(-inf) == 0x007FFFFF  (used as init pattern)

// ---------------------------------------------------------------------------
// WMMA fragment swizzle maps (wave32 16x16x32 f16, CDNA5 ISA 7.12.2).
// LDS tiles are stored in fragment order ([lane][e], 512 halves per subtile)
// so a fragment load is one contiguous 32-byte vector read per lane.
// Both maps place (k even, k+1) at consecutive e in the same lane -> packed
// 2-half (b32) staging writes.
// ---------------------------------------------------------------------------
__device__ __forceinline__ int a_swz(int row, int k) {
    const int kk = k & 15;
    const int lane = row + ((kk & 8) ? 16 : 0);
    const int v = ((kk & 7) >> 1) + ((k & 16) ? 4 : 0);
    return lane * 16 + 2 * v + (k & 1);
}
__device__ __forceinline__ int b_swz(int k, int col) {
    const int lane = col + ((k & 8) ? 16 : 0);
    return lane * 16 + (k & 7) + ((k & 16) ? 8 : 0);
}

__device__ __forceinline__ v16h frag_ld(const _Float16* p) {
    return *(const v16h*)(p + (threadIdx.x & 31) * 16);
}

// ---------------------------------------------------------------------------
// build q = concat(x[:,512:], landmarks@Wp+bp, c@Ws+bs)  -> f16 [N,640]
// ---------------------------------------------------------------------------
__global__ __launch_bounds__(256) void build_q_kernel(
    const float* __restrict__ x, const float* __restrict__ lm,
    const float* __restrict__ cc, const float* __restrict__ Wp,
    const float* __restrict__ bp, const float* __restrict__ Ws,
    const float* __restrict__ bs, _Float16* __restrict__ qh) {
    const int row = blockIdx.x;
    for (int j = threadIdx.x; j < EMB; j += 256) {
        float v;
        if (j < FD / 2) {
            v = x[(size_t)row * FD + FD / 2 + j];
        } else if (j < FD / 2 + PROJ) {
            const int o = j - FD / 2;
            v = bp[o];
            const float* l = lm + (size_t)row * DL;
            for (int k = 0; k < DL; ++k) v += l[k] * Wp[k * PROJ + o];
        } else {
            const int o = j - FD / 2 - PROJ;
            v = bs[o];
            const float* l = cc + (size_t)row * DC;
            for (int k = 0; k < DC; ++k) v += l[k] * Ws[k * PROJ + o];
        }
        qh[(size_t)row * EMB + j] = (_Float16)v;
    }
}

// ---------------------------------------------------------------------------
// Generic WMMA GEMM: C = alpha*(A@B) + bias
//   REQUIRES: M%64==0, Nc%64==0, K%32==0 (true for every call site here).
//   A: f16 [M,K] row-major.  BMODE: 0 = f32 B[k,n], 1 = f16 B[n,k] (trans),
//   2 = f16 B[k,n].  128 threads (4 waves), 64x64 block tile, double-buffered
//   fragment-swizzled LDS.
// ---------------------------------------------------------------------------
template <int BMODE>
__global__ __launch_bounds__(128) void gemm_wmma_kernel(
    const _Float16* __restrict__ A, const void* __restrict__ Bv, int ldb,
    const float* __restrict__ bias, float alpha, float* __restrict__ Cf,
    _Float16* __restrict__ Ch, int M, int Nc, int K) {
    __shared__ alignas(32) _Float16 As[2][4][512];   // [buf][row subtile]
    __shared__ alignas(32) _Float16 Bs[2][4][512];   // [buf][col subtile]
    const int tid = threadIdx.x, lane = tid & 31, wid = tid >> 5;
    const int row0 = blockIdx.x * 64, col0 = blockIdx.y * 64;
    const float*    Bf = (const float*)Bv;
    const _Float16* Bh = (const _Float16*)Bv;

    auto stage = [&](int pb, int k0) {
        // A: 64x32 halves = 1024 packed pairs, swizzled
#pragma unroll
        for (int i = tid; i < 1024; i += 128) {
            const int r = i >> 4;            // 0..63
            const int k = (i & 15) * 2;      // 0,2,..,30
            const unsigned pk =
                *(const unsigned*)(A + (size_t)(row0 + r) * K + k0 + k);
            *(unsigned*)&As[pb][r >> 4][a_swz(r & 15, k)] = pk;
        }
        // B: 32x64 halves = 1024 packed k-pairs, swizzled
#pragma unroll
        for (int i = tid; i < 1024; i += 128) {
            const int c = i & 63;            // 0..63
            const int k = (i >> 6) * 2;      // 0,2,..,30
            const int gk = k0 + k, gn = col0 + c;
            hpair pk;
            if (BMODE == 0) {                // f32, B[k,n]
                pk.h[0] = (_Float16)Bf[(size_t)gk * ldb + gn];
                pk.h[1] = (_Float16)Bf[(size_t)(gk + 1) * ldb + gn];
            } else if (BMODE == 1) {         // f16, transposed: B[n,k]
                pk.u = *(const unsigned*)(Bh + (size_t)gn * ldb + gk);
            } else {                         // f16, B[k,n]
                pk.h[0] = Bh[(size_t)gk * ldb + gn];
                pk.h[1] = Bh[(size_t)(gk + 1) * ldb + gn];
            }
            *(unsigned*)&Bs[pb][c >> 4][b_swz(k, c & 15)] = pk.u;
        }
    };

    v8f z = {};
    v8f acc[4] = {z, z, z, z};

    stage(0, 0);
    __syncthreads();
    int pb = 0;
    for (int k0 = 0; k0 < K; k0 += 32) {
        if (k0 + 32 < K) stage(pb ^ 1, k0 + 32);   // prefetch next tile
        const v16h a = frag_ld(As[pb][wid]);
#pragma unroll
        for (int nt = 0; nt < 4; ++nt) {
            const v16h b = frag_ld(Bs[pb][nt]);
            acc[nt] = __builtin_amdgcn_wmma_f32_16x16x32_f16(
                false, a, false, b, (short)0, acc[nt], false, false);
        }
        __syncthreads();                           // reads done + next staged
        pb ^= 1;
    }

    // C/D layout: lane l -> col = l&15 ; vgpr r -> row = r + 8*(l>=16)
    const int rr_off = (lane >> 4) * 8, cc = lane & 15;
#pragma unroll
    for (int nt = 0; nt < 4; ++nt) {
        const int gn = col0 + nt * 16 + cc;
        const float bval = bias ? bias[gn] : 0.0f;
#pragma unroll
        for (int r = 0; r < 8; ++r) {
            const int gm = row0 + wid * 16 + rr_off + r;
            const float v = acc[nt][r] * alpha + bval;
            if (Cf) Cf[(size_t)gm * Nc + gn] = v;
            if (Ch) Ch[(size_t)gm * Nc + gn] = (_Float16)v;
        }
    }
}

// ---------------------------------------------------------------------------
// Row softmax of S [N,N] f32 -> A [N,N] f16
// ---------------------------------------------------------------------------
__global__ __launch_bounds__(256) void softmax_rows_kernel(
    const float* __restrict__ S, _Float16* __restrict__ Ah, int Ncols) {
    __shared__ float red[256];
    const int row = blockIdx.x, tid = threadIdx.x;
    const float* sr = S + (size_t)row * Ncols;
    float m = -3.4e38f;
    for (int j = tid; j < Ncols; j += 256) m = fmaxf(m, sr[j]);
    red[tid] = m; __syncthreads();
    for (int s = 128; s > 0; s >>= 1) {
        if (tid < s) red[tid] = fmaxf(red[tid], red[tid + s]);
        __syncthreads();
    }
    m = red[0]; __syncthreads();
    float sum = 0.0f;
    for (int j = tid; j < Ncols; j += 256) sum += __expf(sr[j] - m);
    red[tid] = sum; __syncthreads();
    for (int s = 128; s > 0; s >>= 1) {
        if (tid < s) red[tid] += red[tid + s];
        __syncthreads();
    }
    const float inv = 1.0f / red[0];
    _Float16* ar = Ah + (size_t)row * Ncols;
    for (int j = tid; j < Ncols; j += 256)
        ar[j] = (_Float16)(__expf(sr[j] - m) * inv);
}

// ---------------------------------------------------------------------------
// Column BatchNorm (biased var, batch stats) + ReLU. One block per column.
// ---------------------------------------------------------------------------
__global__ __launch_bounds__(256) void bn_relu_col_kernel(
    const float* __restrict__ X, int M, int C, const float* __restrict__ g,
    const float* __restrict__ b, float* __restrict__ Y,
    _Float16* __restrict__ Yh, int accumulate) {
    __shared__ float r1[256], r2[256];
    const int c = blockIdx.x, tid = threadIdx.x;
    float s1 = 0.0f, s2 = 0.0f;
    for (int r = tid; r < M; r += 256) {
        const float v = X[(size_t)r * C + c];
        s1 += v; s2 += v * v;
    }
    r1[tid] = s1; r2[tid] = s2; __syncthreads();
    for (int s = 128; s > 0; s >>= 1) {
        if (tid < s) { r1[tid] += r1[tid + s]; r2[tid] += r2[tid + s]; }
        __syncthreads();
    }
    const float mean = r1[0] / (float)M;
    const float var  = r2[0] / (float)M - mean * mean;
    const float inv  = rsqrtf(var + 1e-5f);
    const float gg = g[c], bb = b[c];
    for (int r = tid; r < M; r += 256) {
        float v = gg * (X[(size_t)r * C + c] - mean) * inv + bb;
        v = fmaxf(v, 0.0f);
        const size_t idx = (size_t)r * C + c;
        if (Y) { if (accumulate) Y[idx] += v; else Y[idx] = v; }
        if (Yh) Yh[idx] = (_Float16)v;
    }
}

// ---------------------------------------------------------------------------
// Fused EdgeConv with WMMA.  128 threads = 4 waves; wave handles one 16-edge
// tile: A = [x_i | x_j - x_i] (16x128 f16) -> relu(A@W1+b1) -> (@W2+b2)
// -> masked atomic-max scatter into aggEnc[dst].  Fragment-swizzled LDS.
// ---------------------------------------------------------------------------
__device__ __forceinline__ bool edge_ok(int attr, int mode) {
    return mode == 2 || (mode == 0 ? (attr <= 0) : (attr >= 0));
}

__global__ __launch_bounds__(128) void edgeconv_wmma_kernel(
    const _Float16* __restrict__ hh, const int* __restrict__ src,
    const int* __restrict__ dst, const int* __restrict__ attr, int mode,
    const float* __restrict__ W1, const float* __restrict__ b1,
    const float* __restrict__ W2, const float* __restrict__ b2,
    unsigned* __restrict__ aggEnc) {
    __shared__ alignas(32) _Float16 W1sw[16][512];    // [ks*4+nt] 32x16 frags
    __shared__ alignas(32) _Float16 W2sw[8][512];     // [ks*4+nt]
    __shared__ alignas(32) _Float16 Asw[4][4][512];   // [wave][ks] 16x32 frags
    __shared__ alignas(32) _Float16 Hsw[4][2][512];   // [wave][ks]

    const int tid = threadIdx.x, lane = tid & 31, wid = tid >> 5;
    const int tile = blockIdx.x * 4 + wid;
    const int ebase = tile * 16;

    // ---- stage weights (packed k-pairs, swizzled) ------------------------
#pragma unroll
    for (int i = tid; i < 4096; i += 128) {          // W1: 128x64, 4096 pairs
        const int n = i & 63, k = (i >> 6) * 2;
        hpair pk;
        pk.h[0] = (_Float16)W1[(size_t)k * 64 + n];
        pk.h[1] = (_Float16)W1[(size_t)(k + 1) * 64 + n];
        *(unsigned*)&W1sw[(k >> 5) * 4 + (n >> 4)][b_swz(k & 31, n & 15)] = pk.u;
    }
#pragma unroll
    for (int i = tid; i < 2048; i += 128) {          // W2: 64x64, 2048 pairs
        const int n = i & 63, k = (i >> 5) * 2;
        hpair pk;
        pk.h[0] = (_Float16)W2[(size_t)k * 64 + n];
        pk.h[1] = (_Float16)W2[(size_t)(k + 1) * 64 + n];
        *(unsigned*)&W2sw[(k >> 5) * 4 + (n >> 4)][b_swz(k & 31, n & 15)] = pk.u;
    }

    // ---- stage per-wave edge tile A = [x_i | x_j - x_i], swizzled --------
#pragma unroll
    for (int p = lane; p < 1024; p += 32) {          // 16x128 halves = 1024 pairs
        const int e = p >> 6;                        // edge row 0..15
        const int k = (p & 63) * 2;                  // 0,2,..,126
        const int eg = ebase + e;
        const int s = src[eg], d = dst[eg];
        hpair out;
        if (k < 64) {                                // x_i block
            out.u = *(const unsigned*)(hh + (size_t)d * C1 + k);
        } else {                                     // x_j - x_i block
            const int kk = k - 64;
            hpair xi, xj;
            xi.u = *(const unsigned*)(hh + (size_t)d * C1 + kk);
            xj.u = *(const unsigned*)(hh + (size_t)s * C1 + kk);
            out.h[0] = (_Float16)((float)xj.h[0] - (float)xi.h[0]);
            out.h[1] = (_Float16)((float)xj.h[1] - (float)xi.h[1]);
        }
        *(unsigned*)&Asw[wid][k >> 5][a_swz(e, k & 31)] = out.u;
    }
    __syncthreads();

    // ---- layer 1: relu(A@W1 + b1) -> Hsw (A-swizzled) --------------------
    v8f z = {};
    v8f acc[4] = {z, z, z, z};
#pragma unroll
    for (int ks = 0; ks < 4; ++ks) {
        const v16h a = frag_ld(Asw[wid][ks]);
#pragma unroll
        for (int nt = 0; nt < 4; ++nt) {
            const v16h b = frag_ld(W1sw[ks * 4 + nt]);
            acc[nt] = __builtin_amdgcn_wmma_f32_16x16x32_f16(
                false, a, false, b, (short)0, acc[nt], false, false);
        }
    }
    const int rr_off = (lane >> 4) * 8, cc = lane & 15;
#pragma unroll
    for (int nt = 0; nt < 4; ++nt) {
        const int col = nt * 16 + cc;
        const float bb = b1[col];
#pragma unroll
        for (int r = 0; r < 8; ++r) {
            const int row = rr_off + r;
            Hsw[wid][col >> 5][a_swz(row, col & 31)] =
                (_Float16)fmaxf(acc[nt][r] + bb, 0.0f);
        }
    }
    __syncthreads();

    // ---- layer 2: H@W2 + b2, masked atomic-max scatter -------------------
    v8f acc2[4] = {z, z, z, z};
#pragma unroll
    for (int ks = 0; ks < 2; ++ks) {
        const v16h a = frag_ld(Hsw[wid][ks]);
#pragma unroll
        for (int nt = 0; nt < 4; ++nt) {
            const v16h b = frag_ld(W2sw[ks * 4 + nt]);
            acc2[nt] = __builtin_amdgcn_wmma_f32_16x16x32_f16(
                false, a, false, b, (short)0, acc2[nt], false, false);
        }
    }
    // hoist per-row edge destination / mask (8 rows per lane)
    int  dcache[8];
    bool okc[8];
#pragma unroll
    for (int r = 0; r < 8; ++r) {
        const int eg = ebase + rr_off + r;
        dcache[r] = dst[eg];
        okc[r]    = edge_ok(attr[eg], mode);
    }
#pragma unroll
    for (int nt = 0; nt < 4; ++nt) {
        const int col = nt * 16 + cc;
        const float bb = b2[col];
#pragma unroll
        for (int r = 0; r < 8; ++r) {
            if (!okc[r]) continue;
            const float v = acc2[nt][r] + bb;
            atomicMax(&aggEnc[(size_t)dcache[r] * C1 + col], encF(v));
        }
    }
}

// ---------------------------------------------------------------------------
// Misc small kernels
// ---------------------------------------------------------------------------
__global__ void fill_u32_kernel(unsigned* p, unsigned v, int n) {
    const int i = blockIdx.x * blockDim.x + threadIdx.x;
    if (i < n) p[i] = v;
}

__global__ void decode_agg_kernel(const unsigned* __restrict__ enc,
                                  float* __restrict__ out, int n) {
    const int i = blockIdx.x * blockDim.x + threadIdx.x;
    if (i >= n) return;
    const float f = decF(enc[i]);
    out[i] = (f < -1e37f) ? 0.0f : f;   // -inf (no contributing edge) -> 0
}

__global__ void gat_score_kernel(const float* __restrict__ xl,
                                 const float* __restrict__ xr,
                                 const float* __restrict__ att,
                                 const int* __restrict__ src,
                                 const int* __restrict__ dst,
                                 const int* __restrict__ attr, int mode,
                                 float* __restrict__ esc,
                                 unsigned* __restrict__ emaxEnc) {
    const int i = blockIdx.x * blockDim.x + threadIdx.x;
    if (i >= E + N) return;
    int s, d; bool ok;
    if (i < E) { s = src[i]; d = dst[i]; ok = edge_ok(attr[i], mode); }
    else       { s = d = i - E; ok = true; }   // self loop
    if (!ok) { esc[i] = -3.4e38f; return; }
    float acc = 0.0f;
#pragma unroll 4
    for (int c = 0; c < C2; ++c) {
        float t = xl[(size_t)s * C2 + c] + xr[(size_t)d * C2 + c];
        t = (t > 0.0f) ? t : 0.2f * t;         // leaky_relu 0.2
        acc += t * att[c];
    }
    esc[i] = acc;
    atomicMax(&emaxEnc[d], encF(acc));
}

__global__ void gat_w_kernel(const float* __restrict__ esc,
                             const unsigned* __restrict__ emaxEnc,
                             const int* __restrict__ dst,
                             float* __restrict__ wbuf,
                             float* __restrict__ denom) {
    const int i = blockIdx.x * blockDim.x + threadIdx.x;
    if (i >= E + N) return;
    const float e = esc[i];
    if (e <= -1e37f) { wbuf[i] = 0.0f; return; }
    const int d = (i < E) ? dst[i] : (i - E);
    const float w = __expf(e - decF(emaxEnc[d]));
    wbuf[i] = w;
    atomicAdd(&denom[d], w);
}

__global__ void gat_accum_kernel(const float* __restrict__ wbuf,
                                 const float* __restrict__ denom,
                                 const float* __restrict__ xl,
                                 const int* __restrict__ src,
                                 const int* __restrict__ dst,
                                 float* __restrict__ outbuf) {
    const int i = blockIdx.x * blockDim.x + threadIdx.x;
    if (i >= E + N) return;
    const float w = wbuf[i];
    if (w == 0.0f) return;
    int s, d;
    if (i < E) { s = src[i]; d = dst[i]; } else { s = d = i - E; }
    const float coef = w / denom[d];
    for (int c = 0; c < C2; ++c)
        atomicAdd(&outbuf[(size_t)d * C2 + c], coef * xl[(size_t)s * C2 + c]);
}

__global__ void final_out_kernel(const float* __restrict__ sum,
                                 const float* __restrict__ Wf,
                                 const float* __restrict__ bf,
                                 float* __restrict__ out) {
    const int i = blockIdx.x * blockDim.x + threadIdx.x;
    if (i >= N) return;
    float acc = bf[0];
#pragma unroll 4
    for (int c = 0; c < C2; ++c) acc += sum[(size_t)i * C2 + c] * Wf[c];
    out[i] = acc;
}

// ---------------------------------------------------------------------------
// Host orchestration
// ---------------------------------------------------------------------------
extern "C" void kernel_launch(void* const* d_in, const int* in_sizes, int n_in,
                              void* d_out, int out_size, void* d_ws,
                              size_t ws_size, hipStream_t stream) {
    (void)in_sizes; (void)n_in; (void)out_size; (void)ws_size;

    // d_in layout follows setup_inputs() dict order (params in insertion order)
    enum { I_X = 0, I_LM, I_C, I_WP, I_BP, I_WS, I_BS, I_WQ, I_BQ, I_WK, I_BK,
           I_WV, I_BV, I_WO, I_BO, I_WPROJ, I_BPROJ, I_G01, I_B01,
           I_BR0 = 19,   // W1_i,b1_i,W2_i,b2_i,g1_i,bn1_i  (x3 branches)
           I_WL = 37, I_WR, I_ATT, I_BG, I_G21, I_B21, I_WF, I_BF,
           I_EI = 45, I_EA = 46 };

    auto F = [&](int i) { return (const float*)d_in[i]; };
    const int* ei   = (const int*)d_in[I_EI];
    const int* src  = ei;
    const int* dst  = ei + E;
    const int* attr = (const int*)d_in[I_EA];

    // workspace bump allocator
    char* base = (char*)d_ws;
    size_t off = 0;
    auto alloc = [&](size_t bytes) -> void* {
        void* p = base + off;
        off += (bytes + 255) & ~(size_t)255;
        return p;
    };

    _Float16* qh   = (_Float16*)alloc((size_t)N * EMB * 2);
    _Float16* Qh   = (_Float16*)alloc((size_t)N * EMB * 2);
    _Float16* Kh   = (_Float16*)alloc((size_t)N * EMB * 2);
    _Float16* Vh   = (_Float16*)alloc((size_t)N * EMB * 2);
    float*    Sbuf = (float*)   alloc((size_t)N * N * 4);
    _Float16* Ah   = (_Float16*)alloc((size_t)N * N * 2);
    _Float16* AVh  = (_Float16*)alloc((size_t)N * EMB * 2);
    _Float16* xvh  = (_Float16*)alloc((size_t)N * EMB * 2);
    float*    hpre = (float*)   alloc((size_t)N * C1 * 4);
    _Float16* hh   = (_Float16*)alloc((size_t)N * C1 * 2);
    unsigned* aggE = (unsigned*)alloc((size_t)N * C1 * 4);
    float*    y0   = (float*)   alloc((size_t)N * C1 * 4);
    _Float16* yh   = (_Float16*)alloc((size_t)N * C1 * 2);
    float*    xl   = (float*)   alloc((size_t)N * C2 * 4);
    float*    xr   = (float*)   alloc((size_t)N * C2 * 4);
    float*    esc  = (float*)   alloc((size_t)(E + N) * 4);
    float*    wbuf = (float*)   alloc((size_t)(E + N) * 4);
    unsigned* emaxE= (unsigned*)alloc((size_t)N * 4);
    float*    denom= (float*)   alloc((size_t)N * 4);
    float*    gout = (float*)   alloc((size_t)N * C2 * 4);
    float*    osum = (float*)   alloc((size_t)N * C2 * 4);

    // BMODE: 0 = f32 B[k,n], 1 = f16 B[n,k] (transposed), 2 = f16 B[k,n]
    auto gemm = [&](const _Float16* A, const void* B, int bmode, int ldb,
                    const float* bias, float alpha, float* Cf, _Float16* Ch,
                    int M_, int N_, int K_) {
        dim3 g(M_ / 64, N_ / 64);
        if (bmode == 0)
            gemm_wmma_kernel<0><<<g, 128, 0, stream>>>(A, B, ldb, bias, alpha,
                                                       Cf, Ch, M_, N_, K_);
        else if (bmode == 1)
            gemm_wmma_kernel<1><<<g, 128, 0, stream>>>(A, B, ldb, bias, alpha,
                                                       Cf, Ch, M_, N_, K_);
        else
            gemm_wmma_kernel<2><<<g, 128, 0, stream>>>(A, B, ldb, bias, alpha,
                                                       Cf, Ch, M_, N_, K_);
    };

    // ---- token assembly + attention -------------------------------------
    build_q_kernel<<<N, 256, 0, stream>>>(F(I_X), F(I_LM), F(I_C), F(I_WP),
                                          F(I_BP), F(I_WS), F(I_BS), qh);

    gemm(qh, F(I_WQ), 0, EMB, F(I_BQ), 1.0f, nullptr, Qh, N, EMB, EMB);
    gemm(qh, F(I_WK), 0, EMB, F(I_BK), 1.0f, nullptr, Kh, N, EMB, EMB);
    gemm(qh, F(I_WV), 0, EMB, F(I_BV), 1.0f, nullptr, Vh, N, EMB, EMB);

    const float scale = 0.039528470752104741f;  // 1/sqrt(640)
    gemm(Qh, Kh, 1, EMB, nullptr, scale, Sbuf, nullptr, N, N, EMB);
    softmax_rows_kernel<<<N, 256, 0, stream>>>(Sbuf, Ah, N);
    gemm(Ah, Vh, 2, EMB, nullptr, 1.0f, nullptr, AVh, N, EMB, N);
    gemm(AVh, F(I_WO), 0, EMB, F(I_BO), 1.0f, nullptr, xvh, N, EMB, EMB);
    // bproj dropped: constant-per-column shift is a no-op under BatchNorm
    gemm(xvh, F(I_WPROJ), 0, C1, nullptr, 1.0f, hpre, nullptr, N, C1, EMB);
    bn_relu_col_kernel<<<C1, 256, 0, stream>>>(hpre, N, C1, F(I_G01), F(I_B01),
                                               nullptr, hh, 0);

    // ---- three masked graph branches ------------------------------------
    const int threads = 256;
    for (int br = 0; br < 3; ++br) {
        const int pb = I_BR0 + br * 6;
        const float *W1 = F(pb), *b1 = F(pb + 1), *W2 = F(pb + 2),
                    *b2 = F(pb + 3), *g1 = F(pb + 4), *bn1 = F(pb + 5);
        const int mode = (br == 0) ? 0 : (br == 1) ? 1 : 2;

        // EdgeConv with max-aggregation
        fill_u32_kernel<<<(N * C1 + threads - 1) / threads, threads, 0, stream>>>(
            aggE, 0x007FFFFFu, N * C1);                      // enc(-inf)
        edgeconv_wmma_kernel<<<E / 64, 128, 0, stream>>>(
            hh, src, dst, attr, mode, W1, b1, W2, b2, aggE);
        decode_agg_kernel<<<(N * C1 + threads - 1) / threads, threads, 0, stream>>>(
            aggE, y0, N * C1);
        bn_relu_col_kernel<<<C1, 256, 0, stream>>>(y0, N, C1, g1, bn1,
                                                   nullptr, yh, 0);

        // GATv2 (shared Wl/Wr/att across branches; bg cancels under BN)
        gemm(yh, F(I_WL), 0, C2, nullptr, 1.0f, xl, nullptr, N, C2, C1);
        gemm(yh, F(I_WR), 0, C2, nullptr, 1.0f, xr, nullptr, N, C2, C1);

        fill_u32_kernel<<<(N + threads - 1) / threads, threads, 0, stream>>>(
            emaxE, 0x007FFFFFu, N);
        fill_u32_kernel<<<(N + threads - 1) / threads, threads, 0, stream>>>(
            (unsigned*)denom, 0u, N);
        fill_u32_kernel<<<(N * C2 + threads - 1) / threads, threads, 0, stream>>>(
            (unsigned*)gout, 0u, N * C2);

        const int tot = E + N;
        gat_score_kernel<<<(tot + threads - 1) / threads, threads, 0, stream>>>(
            xl, xr, F(I_ATT), src, dst, attr, mode, esc, emaxE);
        gat_w_kernel<<<(tot + threads - 1) / threads, threads, 0, stream>>>(
            esc, emaxE, dst, wbuf, denom);
        gat_accum_kernel<<<(tot + threads - 1) / threads, threads, 0, stream>>>(
            wbuf, denom, xl, src, dst, gout);

        bn_relu_col_kernel<<<C2, 256, 0, stream>>>(gout, N, C2, F(I_G21),
                                                   F(I_B21), osum, nullptr,
                                                   br > 0 ? 1 : 0);
    }

    // ---- final linear [N,64] @ [64,1] + bf ------------------------------
    final_out_kernel<<<(N + threads - 1) / threads, threads, 0, stream>>>(
        osum, F(I_WF), F(I_BF), (float*)d_out);
}
